// LocationSensitiveAttentionLayer_88502096101492
// MI455X (gfx1250) — matrix operand order repaired and checked by reading, not compiled, not run on hardware
//
#include <hip/hip_runtime.h>
#include <hip/hip_bf16.h>
#include <math.h>

// ---------------- types ----------------
typedef __bf16 bf16t;
typedef bf16t v16bf __attribute__((ext_vector_type(16)));
typedef bf16t v8bf  __attribute__((ext_vector_type(8)));
typedef float v8f   __attribute__((ext_vector_type(8)));
typedef float v4f   __attribute__((ext_vector_type(4)));

// ---------------- problem constants ----------------
#define B_SZ   32
#define T_ENC  400
#define ENC_D  512
#define T_DEC  800
#define DEC_D  80
#define UNITS  128
#define FILTERS 32
#define KSZ    31
#define LSTM   1024
#define NG     4096        // 4*LSTM
#define K0     1616        // DEC_D + ENC_D + LSTM
#define NKT0   51          // padded K = 1632
#define NKT1   64          // 2048/32

__device__ __forceinline__ float sigf(float x) { return 1.0f / (1.0f + __expf(-x)); }

// ---------------- CDNA5 async copy to LDS (ASYNCcnt-tracked) ----------------
__device__ __forceinline__ void async_copy_b128(unsigned lds_off, const void* gptr) {
    asm volatile("global_load_async_to_lds_b128 %0, %1, off"
                 :: "v"(lds_off), "v"(gptr) : "memory");
}
__device__ __forceinline__ void wait_async0() {
    asm volatile("s_wait_asynccnt 0x0" ::: "memory");
}
__device__ __forceinline__ unsigned lds_addr(const void* p) {
    // generic pointers to LDS carry the byte offset in the low 32 bits
    return (unsigned)(size_t)p;
}

// ---------------- WMMA fragment helpers ----------------
// A fragment (16x32 bf16): lane L holds row M = L%16; two contiguous 8-half runs
// at K = kb..kb+7 and kb+16..kb+23, kb = (L<16 ? 0 : 8).
__device__ __forceinline__ v16bf combine_runs(v8bf lo, v8bf hi) {
    v16bf r;
#pragma unroll
    for (int i = 0; i < 8; ++i) { r[i] = lo[i]; r[8 + i] = hi[i]; }
    return r;
}

// A fragment gathered from row-major f32 global memory (keys precompute only).
__device__ __forceinline__ v16bf frag_a_f32(const float* __restrict__ rowbase, int ldk, int lane) {
    int m  = lane & 15;
    int kb = (lane < 16) ? 0 : 8;
    const float* p = rowbase + (size_t)m * ldk + kb;
    v4f x0 = *(const v4f*)(p);
    v4f x1 = *(const v4f*)(p + 4);
    v4f x2 = *(const v4f*)(p + 16);
    v4f x3 = *(const v4f*)(p + 20);
    v16bf r;
#pragma unroll
    for (int i = 0; i < 4; ++i) {
        r[i]      = (bf16t)x0[i];
        r[4 + i]  = (bf16t)x1[i];
        r[8 + i]  = (bf16t)x2[i];
        r[12 + i] = (bf16t)x3[i];
    }
    return r;
}

// ---------------- weight packing ----------------
// B fragment (32x16 bf16): lane n holds K=0..7,16..23 for column n, lane n+16 holds K=8..15,24..31.
// Packed: dst[((nt*nkt)+kt)*512 + lane*16 + e]; LSTM columns gate-interleaved:
// nt -> j=nt/4, g=nt%4, original column = g*1024 + j*16 + n. Rows are [W ; U ; 0-pad].
__global__ void pack_gate(const float* __restrict__ W, const float* __restrict__ U,
                          bf16t* __restrict__ dst, int kW, int kU, int nkt) {
    size_t idx   = (size_t)blockIdx.x * 256 + threadIdx.x;
    size_t total = (size_t)256 * nkt * 512;
    if (idx >= total) return;
    int    e    = (int)(idx & 15);
    int    lane = (int)((idx >> 4) & 31);
    size_t tile = idx >> 9;
    int    kt   = (int)(tile % (size_t)nkt);
    int    nt   = (int)(tile / (size_t)nkt);
    int    n    = lane & 15;
    int    k    = kt * 32 + ((lane < 16) ? 0 : 8) + ((e < 8) ? e : 8 + e);
    int    jj   = nt >> 2, g = nt & 3;
    int    col  = g * LSTM + jj * 16 + n;
    float  v = 0.0f;
    if (k < kW)            v = W[(size_t)k * NG + col];
    else if (k < kW + kU)  v = U[(size_t)(k - kW) * NG + col];
    dst[idx] = (bf16t)v;
}

__global__ void pack_plain(const float* __restrict__ W, bf16t* __restrict__ dst,
                           int N, int nkt, int ntiles) {
    size_t idx   = (size_t)blockIdx.x * 256 + threadIdx.x;
    size_t total = (size_t)ntiles * nkt * 512;
    if (idx >= total) return;
    int    e    = (int)(idx & 15);
    int    lane = (int)((idx >> 4) & 31);
    size_t tile = idx >> 9;
    int    kt   = (int)(tile % (size_t)nkt);
    int    nt   = (int)(tile / (size_t)nkt);
    int    n    = lane & 15;
    int    k    = kt * 32 + ((lane < 16) ? 0 : 8) + ((e < 8) ? e : 8 + e);
    int    col  = nt * 16 + n;
    dst[idx] = (bf16t)W[(size_t)k * N + col];
}

__global__ void zero_kernel(float* __restrict__ p, int n) {
    int i = blockIdx.x * 256 + threadIdx.x;
    if (i < n) p[i] = 0.0f;
}

__global__ void cvt_bf16_kernel(const float* __restrict__ src, bf16t* __restrict__ dst, int n) {
    int i = blockIdx.x * 256 + threadIdx.x;
    if (i < n) dst[i] = (bf16t)src[i];
}

// ---------------- keys = values @ Wm  (M=12800, N=128, K=512) ----------------
__global__ __launch_bounds__(256) void keys_kernel(const float* __restrict__ values,
                                                   const bf16t* __restrict__ WmP,
                                                   float* __restrict__ keys) {
    const int lane = threadIdx.x & 31;
    const int nt   = threadIdx.x >> 5;
    const int mt   = blockIdx.x;
    const float* abase = values + (size_t)mt * 16 * ENC_D;
    v8f acc = {0.f, 0.f, 0.f, 0.f, 0.f, 0.f, 0.f, 0.f};
#pragma unroll 4
    for (int kt = 0; kt < 16; ++kt) {
        v16bf a = frag_a_f32(abase + kt * 32, ENC_D, lane);
        v16bf b = *(const v16bf*)(WmP + (size_t)(nt * 16 + kt) * 512 + lane * 16);
        acc = __builtin_amdgcn_wmma_f32_16x16x32_bf16(false, a, false, b, (short)0, acc, false, false);
    }
    int col   = nt * 16 + (lane & 15);
    int rbase = mt * 16 + ((lane < 16) ? 0 : 8);
#pragma unroll
    for (int r = 0; r < 8; ++r) keys[(size_t)(rbase + r) * UNITS + col] = acc[r];
}

// ---------------- fused WMMA LSTM layer ----------------
// All recurrent activations are kept in bf16, so each lane loads its A-fragment runs
// directly from global memory (two b128 loads) -- no LDS staging, no per-chunk barrier.
// Grid: 64 workgroups (j = 16-column slice of each gate), 8 waves: gate g = w&3, rt = w>>2.
template <int LAYER>
__global__ __launch_bounds__(256) void lstm_kernel(
    const bf16t* __restrict__ Wp, const float* __restrict__ bias,
    const bf16t* __restrict__ qbf,    // layer0: bf16 decoder queries
    const bf16t* __restrict__ ctxbf,  // layer0: bf16 context from previous step
    const bf16t* __restrict__ hA,     // layer0: h0_prev ; layer1: h0_new
    const bf16t* __restrict__ hB,     // layer1: h1_prev
    const float* __restrict__ cprev, float* __restrict__ cnew,
    bf16t* __restrict__ hnew, int t) {
    const int nkt = (LAYER == 0) ? NKT0 : NKT1;
    __shared__ float zt[4][32 * 16];

    const int tid  = threadIdx.x;
    const int lane = tid & 31, wave = tid >> 5;
    const int g = wave & 3, rt = wave >> 2;
    const int j = blockIdx.x;
    const bf16t* wbase = Wp + (size_t)(j * 4 + g) * nkt * 512;

    const int m  = rt * 16 + (lane & 15);
    const int kb = (lane < 16) ? 0 : 8;

    // load one 8-half run of the concatenated activation vector at offset k for row m
    auto load_run = [&](int k) -> v8bf {
        if (LAYER == 0) {
            if (k < DEC_D)          return *(const v8bf*)(qbf + ((size_t)m * T_DEC + t) * DEC_D + k);
            if (k < DEC_D + ENC_D)  return *(const v8bf*)(ctxbf + m * ENC_D + (k - DEC_D));
            if (k < K0)             return *(const v8bf*)(hA + m * LSTM + (k - (DEC_D + ENC_D)));
            v8bf z;
#pragma unroll
            for (int i = 0; i < 8; ++i) z[i] = (bf16t)0.0f;
            return z;
        } else {
            if (k < LSTM) return *(const v8bf*)(hA + m * LSTM + k);
            return *(const v8bf*)(hB + m * LSTM + (k - LSTM));
        }
    };

    v8f acc = {0.f, 0.f, 0.f, 0.f, 0.f, 0.f, 0.f, 0.f};
#pragma unroll 2
    for (int kt = 0; kt < nkt; ++kt) {
        if (kt + 4 < nkt) __builtin_prefetch(wbase + (size_t)(kt + 4) * 512, 0, 0);
        int kbase = kt * 32 + kb;
        v16bf a = combine_runs(load_run(kbase), load_run(kbase + 16));
        v16bf b = *(const v16bf*)(wbase + (size_t)kt * 512 + lane * 16);
        acc = __builtin_amdgcn_wmma_f32_16x16x32_bf16(false, a, false, b, (short)0, acc, false, false);
    }

    // exchange gate tiles through LDS (single barrier)
    {
        int col = lane & 15;
        int rb  = (lane < 16) ? 0 : 8;
#pragma unroll
        for (int r = 0; r < 8; ++r) zt[g][(rt * 16 + rb + r) * 16 + col] = acc[r];
    }
    __syncthreads();

    // fused LSTM elementwise: 512 outputs, 2 per thread
#pragma unroll
    for (int ii = 0; ii < 2; ++ii) {
        int idx = tid + ii * 256;
        int row = idx >> 4, col = idx & 15;
        int ocol = j * 16 + col;
        float zi = zt[0][row * 16 + col] + bias[0 * LSTM + ocol];
        float zf = zt[1][row * 16 + col] + bias[1 * LSTM + ocol];
        float zg = zt[2][row * 16 + col] + bias[2 * LSTM + ocol];
        float zo = zt[3][row * 16 + col] + bias[3 * LSTM + ocol];
        float cp = cprev[row * LSTM + ocol];
        float cn = sigf(zf) * cp + sigf(zi) * tanhf(zg);
        float hn = sigf(zo) * tanhf(cn);
        cnew[row * LSTM + ocol] = cn;
        hnew[row * LSTM + ocol] = (bf16t)hn;
    }
}

// ---------------- attention step (one block per batch element) ----------------
__global__ __launch_bounds__(256) void attention_kernel(
    const float* __restrict__ values, const float* __restrict__ keys,
    const bf16t* __restrict__ h1bf, const float* __restrict__ Wq,
    const float* __restrict__ convk, const float* __restrict__ convb,
    const float* __restrict__ Wloc, const float* __restrict__ va,
    const float* __restrict__ ba, float* __restrict__ align_cum,
    bf16t* __restrict__ ctx_bf,
    float* __restrict__ out_c, float* __restrict__ out_e, int t) {
    const int b = blockIdx.x, tid = threadIdx.x;
    __shared__ float s_h1[LSTM];
    __shared__ float s_align[T_ENC + 2 * 15];
    __shared__ float s_pq[UNITS];
    __shared__ __align__(16) float s_wloc[FILTERS * UNITS];
    __shared__ __align__(16) float s_ck[KSZ * FILTERS];
    __shared__ float s_cb[FILTERS];
    __shared__ float s_va[UNITS];
    __shared__ float s_e[T_ENC];
    __shared__ float red[256];

    // ---- async-stage the big read-only tables straight into LDS (ASYNCcnt) ----
#pragma unroll
    for (int i = tid * 4; i < FILTERS * UNITS; i += 256 * 4)
        async_copy_b128(lds_addr(&s_wloc[i]), Wloc + i);
    if (tid < (KSZ * FILTERS) / 4)
        async_copy_b128(lds_addr(&s_ck[tid * 4]), convk + tid * 4);

    // ---- regular staging (conversions / small tables) overlaps the async copies ----
    for (int i = tid; i < LSTM; i += 256) s_h1[i] = (float)h1bf[b * LSTM + i];
    for (int i = tid; i < T_ENC; i += 256) s_align[15 + i] = align_cum[b * T_ENC + i];
    if (tid < 15) { s_align[tid] = 0.0f; s_align[T_ENC + 15 + tid] = 0.0f; }
    if (tid < FILTERS) s_cb[tid] = convb[tid];
    if (tid < UNITS)   s_va[tid] = va[tid];
    wait_async0();
    __syncthreads();

    // pq = h1 @ Wq + b_a, split-K over 2 half-sums
    {
        int u = tid & 127, hh = tid >> 7;
        float pp = 0.0f;
        for (int k = hh * 512; k < hh * 512 + 512; ++k)
            pp += s_h1[k] * Wq[(size_t)k * UNITS + u];
        red[tid] = pp;
        __syncthreads();
        if (tid < UNITS) s_pq[tid] = red[tid] + red[tid + 128] + ba[tid];
        __syncthreads();
    }

    // energies: conv(align) -> loc projection -> tanh reduction
    for (int p = tid; p < T_ENC; p += 256) {
        float feat[FILTERS];
#pragma unroll
        for (int f = 0; f < FILTERS; ++f) feat[f] = s_cb[f];
        for (int dk = 0; dk < KSZ; ++dk) {
            float av = s_align[p + dk];
#pragma unroll
            for (int f = 0; f < FILTERS; ++f) feat[f] += av * s_ck[dk * FILTERS + f];
        }
        float e = 0.0f;
        const float* krow = keys + ((size_t)b * T_ENC + p) * UNITS;
        for (int u = 0; u < UNITS; ++u) {
            float loc = 0.0f;
#pragma unroll
            for (int f = 0; f < FILTERS; ++f) loc += feat[f] * s_wloc[f * UNITS + u];
            e += s_va[u] * tanhf(krow[u] + s_pq[u] + loc);
        }
        s_e[p] = e;
    }
    __syncthreads();

    // softmax over T_enc
    float mx = -1e30f;
    for (int i = tid; i < T_ENC; i += 256) mx = fmaxf(mx, s_e[i]);
    red[tid] = mx; __syncthreads();
    for (int s = 128; s > 0; s >>= 1) { if (tid < s) red[tid] = fmaxf(red[tid], red[tid + s]); __syncthreads(); }
    mx = red[0];
    __syncthreads();
    float sm = 0.0f;
    for (int i = tid; i < T_ENC; i += 256) { float ex = __expf(s_e[i] - mx); s_e[i] = ex; sm += ex; }
    red[tid] = sm; __syncthreads();
    for (int s = 128; s > 0; s >>= 1) { if (tid < s) red[tid] += red[tid + s]; __syncthreads(); }
    float inv = 1.0f / red[0];
    __syncthreads();

    // normalize, emit e_outputs, update cumulative alignment
    for (int i = tid; i < T_ENC; i += 256) {
        float a = s_e[i] * inv;
        s_e[i] = a;
        out_e[((size_t)b * T_DEC + t) * T_ENC + i] = a;
        align_cum[b * T_ENC + i] = s_align[15 + i] + a;
    }
    __syncthreads();

    // context = a @ values -> c_outputs[b,t,:] directly (== final einsum row) + bf16 mirror
    float c0a = 0.0f, c1a = 0.0f;
    for (int p = 0; p < T_ENC; ++p) {
        float av = s_e[p];
        const float* vr = values + ((size_t)b * T_ENC + p) * ENC_D;
        c0a += av * vr[tid];
        c1a += av * vr[tid + 256];
    }
    out_c[((size_t)b * T_DEC + t) * ENC_D + tid]       = c0a;
    out_c[((size_t)b * T_DEC + t) * ENC_D + tid + 256] = c1a;
    ctx_bf[b * ENC_D + tid]       = (bf16t)c0a;
    ctx_bf[b * ENC_D + tid + 256] = (bf16t)c1a;
}

// ---------------- host launcher ----------------
extern "C" void kernel_launch(void* const* d_in, const int* in_sizes, int n_in,
                              void* d_out, int out_size, void* d_ws, size_t ws_size,
                              hipStream_t stream) {
    (void)in_sizes; (void)n_in; (void)out_size; (void)ws_size;
    const float* enc   = (const float*)d_in[0];
    const float* dec   = (const float*)d_in[1];
    const float* Wm    = (const float*)d_in[2];
    const float* Wq    = (const float*)d_in[3];
    const float* convk = (const float*)d_in[4];
    const float* convb = (const float*)d_in[5];
    const float* Wloc  = (const float*)d_in[6];
    const float* va    = (const float*)d_in[7];
    const float* ba    = (const float*)d_in[8];
    const float* W0    = (const float*)d_in[9];
    const float* U0    = (const float*)d_in[10];
    const float* b0    = (const float*)d_in[11];
    const float* W1    = (const float*)d_in[12];
    const float* U1    = (const float*)d_in[13];
    const float* b1    = (const float*)d_in[14];

    float* out_c = (float*)d_out;
    float* out_e = out_c + (size_t)B_SZ * T_DEC * ENC_D;

    char*  ws  = (char*)d_ws;
    size_t off = 0;
    auto take = [&](size_t bytes) -> char* {
        char* p = ws + off;
        off = (off + bytes + 255) & ~(size_t)255;
        return p;
    };
    bf16t* w0p   = (bf16t*)take((size_t)256 * NKT0 * 512 * sizeof(bf16t));
    bf16t* w1p   = (bf16t*)take((size_t)256 * NKT1 * 512 * sizeof(bf16t));
    bf16t* wmp   = (bf16t*)take((size_t)8 * 16 * 512 * sizeof(bf16t));
    float* keysw = (float*)take((size_t)B_SZ * T_ENC * UNITS * sizeof(float));
    bf16t* qbf   = (bf16t*)take((size_t)B_SZ * T_DEC * DEC_D * sizeof(bf16t));
    // bf16 state block: 4 x h (32x1024) + context mirror (32x512)
    const int nbf = 4 * B_SZ * LSTM + B_SZ * ENC_D;
    bf16t* bfst  = (bf16t*)take((size_t)nbf * sizeof(bf16t));
    bf16t* hb0a = bfst;
    bf16t* hb0b = hb0a + B_SZ * LSTM;
    bf16t* hb1a = hb0b + B_SZ * LSTM;
    bf16t* hb1b = hb1a + B_SZ * LSTM;
    bf16t* ctxb = hb1b + B_SZ * LSTM;
    // f32 state block: c0, c1, align_cum
    const int nf = 2 * B_SZ * LSTM + B_SZ * T_ENC;
    float* fst  = (float*)take((size_t)nf * sizeof(float));
    float* c0  = fst;
    float* c1  = c0 + B_SZ * LSTM;
    float* alg = c1 + B_SZ * LSTM;

    { size_t n = (size_t)256 * NKT0 * 512; pack_gate<<<(unsigned)((n + 255) / 256), 256, 0, stream>>>(W0, U0, w0p, 592, 1024, NKT0); }
    { size_t n = (size_t)256 * NKT1 * 512; pack_gate<<<(unsigned)((n + 255) / 256), 256, 0, stream>>>(W1, U1, w1p, 1024, 1024, NKT1); }
    { size_t n = (size_t)8 * 16 * 512;     pack_plain<<<(unsigned)((n + 255) / 256), 256, 0, stream>>>(Wm, wmp, UNITS, 16, 8); }
    { int n = B_SZ * T_DEC * DEC_D;        cvt_bf16_kernel<<<(n + 255) / 256, 256, 0, stream>>>(dec, qbf, n); }
    { int n = nbf / 2;                     zero_kernel<<<(n + 255) / 256, 256, 0, stream>>>((float*)bfst, n); }
    {                                      zero_kernel<<<(nf + 255) / 256, 256, 0, stream>>>(fst, nf); }
    keys_kernel<<<(B_SZ * T_ENC) / 16, 256, 0, stream>>>(enc, wmp, keysw);

    bf16t* h0[2] = {hb0a, hb0b};
    bf16t* h1[2] = {hb1a, hb1b};
    for (int t = 0; t < T_DEC; ++t) {
        const int pp = t & 1, np = pp ^ 1;
        lstm_kernel<0><<<64, 256, 0, stream>>>(w0p, b0, qbf, ctxb, h0[pp], nullptr, c0, c0, h0[np], t);
        lstm_kernel<1><<<64, 256, 0, stream>>>(w1p, b1, nullptr, nullptr, h0[np], h1[pp], c1, c1, h1[np], t);
        attention_kernel<<<B_SZ, 256, 0, stream>>>(enc, keysw, h1[np], Wq, convk, convb, Wloc,
                                                   va, ba, alg, ctxb, out_c, out_e, t);
    }
}
